// WaveNet_21586505629707
// MI455X (gfx1250) — matrix operand order repaired and checked by reading, not compiled, not run on hardware
//
#include <hip/hip_runtime.h>
#include <hip/hip_bf16.h>

// ---------------- constants (match reference) ----------------
#define LAYERS 30
#define LPS    10
#define KW     3
#define RC     64
#define GC     128
#define SC     64
#define AC     80
#define ACP    96          // AC padded to multiple of 32 for K-tiling
#define BB     8
#define TLEN   8192
#define TT     64          // time tile per workgroup

typedef __attribute__((ext_vector_type(16))) _Float16 v16h;
typedef __attribute__((ext_vector_type(8)))  _Float16 v8h;
typedef __attribute__((ext_vector_type(8)))  float    v8f;

// LDS layout sizes (in _Float16 elements)
#define SZ_CONVW (3 * GC * RC)      // [tap][gc][rc]      24576
#define SZ_AUXW  (GC * ACP)         // [gc][k]            12288
#define SZ_OUTW  (GC * (GC/2))      // [row][k]            8192
#define SZ_XS    (3 * TT * RC)      // [tap][t][ch]       12288
#define SZ_CS    (TT * ACP)         // [t][k]              6144
#define SZ_ZS    (TT * (GC/2))      // [t][ch]             4096
#define SZ_X32   (TT * RC * 2)      // raw f32 center tile (in half units)
#define LDS_HALVES (SZ_CONVW + SZ_AUXW + SZ_OUTW + SZ_XS + SZ_CS + SZ_ZS + SZ_X32)
#define LDS_BYTES  (LDS_HALVES * 2) // 151552 B (< 320KB CDNA5 WGP LDS)

// ---------------- WMMA fragment loaders ----------------
// A-matrix 16x32 f16 layout: lane = 16*half + m,
//   elem j -> K = 16*half + j (contiguous, matches B)
__device__ __forceinline__ v16h load_a_frag(const _Float16* W, int ldk,
                                            int m0, int k0, int lane) {
  const _Float16* p = W + (m0 + (lane & 15)) * ldk + k0 + 16 * (lane >> 4);
  return *(const v16h*)p;           // 32B aligned -> 2x ds_read_b128
}

// B-matrix 32x16 f16 layout: lane = 16*(k>=16) + n, elem j -> K = 16*half + j.
// Activations staged [t][ch] so the 16 K-values per lane are contiguous.
__device__ __forceinline__ v16h load_b_frag(const _Float16* X, int ldc,
                                            int n0, int k0, int lane) {
  const _Float16* p = X + (n0 + (lane & 15)) * ldc + k0 + 16 * (lane >> 4);
  return *(const v16h*)p;           // 32B aligned -> 2x ds_read_b128
}

__device__ __forceinline__ v8f wmma16(v16h a, v16h b, v8f c) {
  return __builtin_amdgcn_wmma_f32_16x16x32_f16(false, a, false, b,
                                                (short)0, c, false, false);
}

// ---------------- kernel 1: first 1x1 conv + zero skips ----------------
__global__ void __launch_bounds__(256) k_init(const float* __restrict__ x,
                                              const float* __restrict__ fw,
                                              const float* __restrict__ fb,
                                              float* __restrict__ x0,
                                              float* __restrict__ skips) {
  int i = blockIdx.x * 256 + threadIdx.x;           // over B*RC*T
  int t  = i & (TLEN - 1);
  int ch = (i >> 13) & (RC - 1);
  int b  = i >> 19;
  x0[i]    = fw[ch] * x[b * TLEN + t] + fb[ch];
  skips[i] = 0.0f;
}

// ---------------- kernel 2: one WaveNet layer (WMMA + async LDS) ----------------
__global__ void __launch_bounds__(256) k_layer(const float* __restrict__ xin,
                                               float* __restrict__ xout,
                                               float* __restrict__ skips,
                                               const float* __restrict__ c,
                                               const float* __restrict__ convw,
                                               const float* __restrict__ convb,
                                               const float* __restrict__ auxw,
                                               const float* __restrict__ outw,
                                               const float* __restrict__ outb,
                                               const float* __restrict__ xmask,
                                               int l, int d) {
  extern __shared__ _Float16 lds[];
  _Float16* convW = lds;                 // [tap][gc][rc]
  _Float16* auxW  = convW + SZ_CONVW;    // [gc][ACP]
  _Float16* outW  = auxW  + SZ_AUXW;     // [row][64]
  _Float16* xs    = outW  + SZ_OUTW;     // [tap][t][ch]
  _Float16* cs    = xs    + SZ_XS;       // [t][ACP]
  _Float16* zs    = cs    + SZ_CS;       // [t][64]
  float*    xs32  = (float*)(zs + SZ_ZS);// raw f32 center tile, [ch][t]

  const int tid = threadIdx.x;
  const int b   = blockIdx.x / (TLEN / TT);
  const int t0  = (blockIdx.x % (TLEN / TT)) * TT;

  // ---- kick off async copy of raw f32 center x tile into LDS ----
  // global_load_async_to_lds_b128 overlaps with the staging work below;
  // tracked on ASYNCcnt (CDNA5 async-to-LDS path).
  {
    const unsigned dsbase = (unsigned)(size_t)xs32;   // low 32b of flat addr = LDS offset
    for (int i = tid * 4; i < TT * RC; i += 256 * 4) {  // i in floats, 4 per op
      int ch = i >> 6;                                  // layout [ch][t], TT=64
      int t  = i & 63;
      unsigned voff = (unsigned)((((size_t)b * RC + ch) * TLEN + t0 + t) * 4u);
      unsigned doff = dsbase + (unsigned)(i * 4);
      asm volatile("global_load_async_to_lds_b128 %0, %1, %2"
                   :: "v"(doff), "v"(voff), "s"(xin) : "memory");
    }
  }

  // ---- stage weights (f32 -> f16), overlapping the async copy ----
  for (int i = tid; i < SZ_CONVW; i += 256) {
    int tap = i / (GC * RC);
    int rem = i - tap * (GC * RC);
    int gc = rem >> 6, rc = rem & 63;
    convW[i] = (_Float16)convw[((l * GC + gc) * RC + rc) * KW + tap];
  }
  for (int i = tid; i < SZ_AUXW; i += 256) {
    int gc = i / ACP, k = i - gc * ACP;
    auxW[i] = (_Float16)(k < AC ? auxw[(l * GC + gc) * AC + k] : 0.0f);
  }
  for (int i = tid; i < SZ_OUTW; i += 256) {
    int gc = i >> 6, k = i & 63;
    outW[i] = (_Float16)outw[(l * GC + gc) * (GC / 2) + k];
  }
  // ---- stage activations: c tile and the two shifted x taps, [t][ch] ----
  for (int i = tid; i < SZ_CS; i += 256) {
    int t = i / ACP, k = i - t * ACP;
    cs[i] = (_Float16)(k < AC ? c[((size_t)b * AC + k) * TLEN + t0 + t] : 0.0f);
  }
#pragma unroll
  for (int tap = 0; tap < 3; tap += 2) {     // taps 0 and 2 from global
    int shift = (tap - 1) * d;
    _Float16* xst = xs + tap * TT * RC;
    for (int i = tid; i < TT * RC; i += 256) {
      int t = i >> 6, ch = i & 63;
      int tg = t0 + t + shift;
      float v = (tg >= 0 && tg < TLEN)
                ? xin[((size_t)b * RC + ch) * TLEN + tg] : 0.0f;
      xst[i] = (_Float16)v;
    }
  }
  // prefetch next layer's conv weights into cache (global_prefetch_b8)
  if (l + 1 < LAYERS) {
    const float* nw = convw + (size_t)(l + 1) * GC * RC * KW;
    for (int i = tid * 16; i < GC * RC * KW; i += 256 * 16)
      __builtin_prefetch(nw + i, 0, 1);
  }

  // async ops of THIS wave retired, then barrier so all waves' copies land
  asm volatile("s_wait_asynccnt 0x0" ::: "memory");
  __syncthreads();

  // ---- center tap (tap 1): convert LDS f32 -> LDS f16 [t][ch] ----
  {
    _Float16* xst = xs + 1 * TT * RC;
    for (int i = tid; i < TT * RC; i += 256) {
      int t = i >> 6, ch = i & 63;
      xst[i] = (_Float16)xs32[ch * TT + t];
    }
  }
  __syncthreads();

  const int lane = tid & 31;
  const int w    = tid >> 5;           // 8 waves

  // ---- phase 1: h = conv(x) + aux(c) + bias, gate -> z (64 x TT) ----
  // wave w: paired M-tiles (xa rows 16*pmt, xb rows 64+16*pmt), 2 N-tiles
  {
    const int pmt   = w & 3;
    const int nbase = (w >> 2) * 2;
    const int mra   = 16 * pmt + 8 * (lane >> 4);   // xa row base for this lane
#pragma unroll
    for (int ni = 0; ni < 2; ++ni) {
      const int nn = nbase + ni;
      v8f acca, accb;
#pragma unroll
      for (int v = 0; v < 8; ++v) {
        acca[v] = convb[l * GC + mra + v];
        accb[v] = convb[l * GC + 64 + mra + v];
      }
#pragma unroll
      for (int tap = 0; tap < 3; ++tap) {
        const _Float16* xst = xs + tap * TT * RC;
#pragma unroll
        for (int kk = 0; kk < 2; ++kk) {
          v16h bf  = load_b_frag(xst, RC, 16 * nn, 32 * kk, lane);
          v16h afa = load_a_frag(convW + tap * GC * RC, RC, 16 * pmt, 32 * kk, lane);
          v16h afb = load_a_frag(convW + tap * GC * RC, RC, 64 + 16 * pmt, 32 * kk, lane);
          acca = wmma16(afa, bf, acca);
          accb = wmma16(afb, bf, accb);
        }
      }
#pragma unroll
      for (int kk = 0; kk < 3; ++kk) {
        v16h bf  = load_b_frag(cs, ACP, 16 * nn, 32 * kk, lane);
        v16h afa = load_a_frag(auxW, ACP, 16 * pmt, 32 * kk, lane);
        v16h afb = load_a_frag(auxW, ACP, 64 + 16 * pmt, 32 * kk, lane);
        acca = wmma16(afa, bf, acca);
        accb = wmma16(afb, bf, accb);
      }
      // gated activation, write z in B-layout [t][ch]
      const int t = 16 * nn + (lane & 15);
#pragma unroll
      for (int v = 0; v < 8; ++v) {
        float za = tanhf(acca[v]);
        float zb = 1.0f / (1.0f + __expf(-accb[v]));
        zs[t * 64 + mra + v] = (_Float16)(za * zb);
      }
    }
  }
  __syncthreads();

  // ---- phase 2: o = out_w @ z + out_b; residual + skip ----
  {
    const int rbase = 16 * w + 8 * (lane >> 4);
#pragma unroll
    for (int nn = 0; nn < 4; ++nn) {
      v8f acc;
#pragma unroll
      for (int v = 0; v < 8; ++v) acc[v] = outb[l * GC + rbase + v];
#pragma unroll
      for (int kk = 0; kk < 2; ++kk) {
        v16h bf = load_b_frag(zs, 64, 16 * nn, 32 * kk, lane);
        v16h af = load_a_frag(outW, 64, 16 * w, 32 * kk, lane);
        acc = wmma16(af, bf, acc);
      }
      const int tl  = 16 * nn + (lane & 15);      // local time
      const int t   = t0 + tl;                    // global time
      const float m = xmask[(size_t)b * TLEN + t];
#pragma unroll
      for (int v = 0; v < 8; ++v) {
        const int r   = rbase + v;
        const float o = acc[v] * m;
        if (w < 4) {                    // residual rows 0..63 (residual from LDS f32)
          size_t idx = ((size_t)b * RC + r) * TLEN + t;
          xout[idx] = xs32[r * TT + tl] + o;
        } else {                        // skip rows 64..127
          size_t idx = ((size_t)b * SC + (r - 64)) * TLEN + t;
          skips[idx] += o;
        }
      }
    }
  }
}

// ---------------- kernel 3: relu -> 1x1 -> relu -> 1x1 ----------------
__global__ void __launch_bounds__(256) k_final(const float* __restrict__ skips,
                                               const float* __restrict__ w1,
                                               const float* __restrict__ b1,
                                               const float* __restrict__ w2,
                                               const float* __restrict__ b2,
                                               float* __restrict__ out) {
  int i = blockIdx.x * 256 + threadIdx.x;     // over B*T
  int b = i >> 13, t = i & (TLEN - 1);
  float s[SC];
#pragma unroll
  for (int ch = 0; ch < SC; ++ch) {
    float v = skips[((size_t)b * SC + ch) * TLEN + t];
    s[ch] = v > 0.0f ? v : 0.0f;
  }
  float y = b2[0];
  for (int o = 0; o < SC; ++o) {
    float a = b1[o];
#pragma unroll
    for (int ch = 0; ch < SC; ++ch) a += w1[o * SC + ch] * s[ch];
    a = a > 0.0f ? a : 0.0f;
    y += w2[o] * a;
  }
  out[i] = y;
}

// ---------------- host launcher ----------------
extern "C" void kernel_launch(void* const* d_in, const int* in_sizes, int n_in,
                              void* d_out, int out_size, void* d_ws, size_t ws_size,
                              hipStream_t stream) {
  (void)in_sizes; (void)n_in; (void)out_size; (void)ws_size;
  const float* x      = (const float*)d_in[0];
  const float* xmask  = (const float*)d_in[1];
  const float* c      = (const float*)d_in[2];
  const float* firstw = (const float*)d_in[3];
  const float* firstb = (const float*)d_in[4];
  const float* convw  = (const float*)d_in[5];
  const float* convb  = (const float*)d_in[6];
  const float* auxw   = (const float*)d_in[7];
  const float* outw   = (const float*)d_in[8];
  const float* outb   = (const float*)d_in[9];
  const float* lw1    = (const float*)d_in[10];
  const float* lb1    = (const float*)d_in[11];
  const float* lw2    = (const float*)d_in[12];
  const float* lb2    = (const float*)d_in[13];

  // workspace: ping-pong residual buffers + skip accumulator (all f32)
  const size_t NXT = (size_t)BB * RC * TLEN;   // 4M elements
  float* xA = (float*)d_ws;
  float* xB = xA + NXT;
  float* sk = xB + NXT;

  // first conv + zero skips
  k_init<<<dim3((unsigned)(NXT / 256)), dim3(256), 0, stream>>>(x, firstw, firstb, xA, sk);

  // 30 layers, ping-pong x buffers
  float* xi = xA; float* xo = xB;
  const dim3 lgrid((unsigned)(BB * (TLEN / TT)));
  for (int l = 0; l < LAYERS; ++l) {
    int d = 1 << (l % LPS);
    k_layer<<<lgrid, dim3(256), LDS_BYTES, stream>>>(xi, xo, sk, c, convw, convb,
                                                     auxw, outw, outb, xmask, l, d);
    float* tmp = xi; xi = xo; xo = tmp;
  }

  // final head
  k_final<<<dim3((BB * TLEN) / 256), dim3(256), 0, stream>>>(sk, lw1, lb1, lw2, lb2,
                                                             (float*)d_out);
}
